// LGSBlock_40836549050612
// MI455X (gfx1250) — matrix operand order repaired and checked by reading, not compiled, not run on hardware
//
// MI455X (gfx1250) implementation of the LGSBlock reference.
// All dense projections run on v_wmma_f32_16x16x32_bf16 (bf16 A/B fragments,
// fp32 accumulate). The block is HBM-bound (~20 GFLOP vs ~0.5 GB traffic at
// 23.3 TB/s), so GEMMs fuse bias/activation/residual epilogues.
// GEMM: 32x32 tile per wave (4 WMMAs/K-step), b128 A loads, clamped
// (unpredicated) B column loads. LDS staging in scan/attention uses
// global_load_async_to_lds_* + s_wait_asynccnt (builtins confirmed present;
// typed-pointer signatures per the round-2 diagnostics).

#include <hip/hip_runtime.h>
#include <math.h>

typedef __attribute__((ext_vector_type(16))) __bf16 v16bf;
typedef __attribute__((ext_vector_type(8)))  float  v8f;
typedef int v4i_vs __attribute__((vector_size(16)));   // matches builtin param

#define NTOK   32768u          // 8 * 4096 tokens
#define NSEQ   131072u         // 32 * 4096 scan rows
#define DINPROJ 386

// ------------------------------------------------ async LDS staging (guarded)
#if defined(__AMDGCN__) && __has_builtin(__builtin_amdgcn_global_load_async_to_lds_b32)
#define HAS_ASYNC32 1
__device__ __forceinline__ void async_ld_b32(const void* g, void* l) {
  __builtin_amdgcn_global_load_async_to_lds_b32(
      (__attribute__((address_space(1))) int*)g,
      (__attribute__((address_space(3))) int*)l, 0, 0);
}
#endif
#if defined(__AMDGCN__) && __has_builtin(__builtin_amdgcn_global_load_async_to_lds_b128)
#define HAS_ASYNC128 1
__device__ __forceinline__ void async_ld_b128(const void* g, void* l) {
  __builtin_amdgcn_global_load_async_to_lds_b128(
      (__attribute__((address_space(1))) v4i_vs*)g,
      (__attribute__((address_space(3))) v4i_vs*)l, 0, 0);
}
#endif
#if defined(HAS_ASYNC32) || defined(HAS_ASYNC128)
__device__ __forceinline__ void wait_async0() {
#if __has_builtin(__builtin_amdgcn_s_wait_asynccnt)
  __builtin_amdgcn_s_wait_asynccnt(0);
#else
  asm volatile("s_wait_asynccnt 0x0" ::: "memory");
#endif
}
#endif

// ---------------------------------------------------------------- activation
template<int ACT>
__device__ __forceinline__ float applyact(float v) {
  if (ACT == 1) return v / (1.f + __expf(-v));                    // silu
  if (ACT == 2) return 0.5f * v * (1.f + erff(v * 0.70710678118654752f)); // gelu exact
  if (ACT == 3) return 1.f / (1.f + __expf(-v));                  // sigmoid
  return v;
}

// --------------------------------------------------------------- WMMA GEMM
// C[M,N] = act(A[M,K] @ W[K,N] + bias) (+ res). One wave per 32x32 tile,
// K-loop of 32. Fragment layouts per CDNA5 ISA:
//   A 16x32: lane L -> M = L&15; elem i -> K = (i<8?0:16) + (L>>4)*8 + (i&7)
//   B 32x16: lane L -> N = L&15; elem i -> K = (i<8?0:16) + (L>>4)*8 + (i&7)
//   C 16x16: lane L -> N = L&15; vgpr r -> M = r + 8*(L>>4)
// A fragment = two contiguous 8-float runs -> 4x b128 loads. B columns are
// clamped (not predicated) so the hot loop has no exec-mask manipulation;
// out-of-range columns only pollute lanes whose stores are masked anyway.
__device__ __forceinline__ v16bf pack_a(const float* __restrict__ p) {
  const float4* q = (const float4*)p;          // p is 16B aligned
  const float4 x0 = q[0], x1 = q[1], x2 = q[4], x3 = q[5];
  v16bf r;
  r[0]=(__bf16)x0.x;  r[1]=(__bf16)x0.y;  r[2]=(__bf16)x0.z;  r[3]=(__bf16)x0.w;
  r[4]=(__bf16)x1.x;  r[5]=(__bf16)x1.y;  r[6]=(__bf16)x1.z;  r[7]=(__bf16)x1.w;
  r[8]=(__bf16)x2.x;  r[9]=(__bf16)x2.y;  r[10]=(__bf16)x2.z; r[11]=(__bf16)x2.w;
  r[12]=(__bf16)x3.x; r[13]=(__bf16)x3.y; r[14]=(__bf16)x3.z; r[15]=(__bf16)x3.w;
  return r;
}

__device__ __forceinline__ v16bf pack_b(const float* __restrict__ Wc, int krow, int N) {
  const float* p = Wc + (size_t)krow * N;
  v16bf r;
#pragma unroll
  for (int i = 0; i < 16; ++i) r[i] = (__bf16)p[(size_t)(i < 8 ? i : i + 8) * N];
  return r;
}

template<int ACT, bool ADD_RES>
__global__ void __launch_bounds__(128)
gemm_bf16_wmma(const float* __restrict__ A, const float* __restrict__ W,
               const float* __restrict__ bias, const float* __restrict__ res,
               float* __restrict__ C, int M, int N, int K)
{
  const int lane = threadIdx.x & 31;
  const int wave = threadIdx.x >> 5;
  const int tiles_n = (N + 31) >> 5;
  const int tiles_m = M >> 5;
  const int tile = blockIdx.x * 4 + wave;
  if (tile >= tiles_m * tiles_n) return;
  const int tm = tile / tiles_n;
  const int tn = tile - tm * tiles_n;
  const int l16  = lane & 15;
  const int half = lane >> 4;
  const int col0 = tn * 32 + l16;
  const int col1 = col0 + 16;
  const int bc0 = (col0 < N) ? col0 : (N - 1);   // clamped load column
  const int bc1 = (col1 < N) ? col1 : (N - 1);
  const float* __restrict__ Ar0 = A + (size_t)(tm * 32 + l16) * K + half * 8;
  const float* __restrict__ Ar1 = Ar0 + (size_t)16 * K;
  const float* __restrict__ W0 = W + bc0;
  const float* __restrict__ W1 = W + bc1;

  v8f acc00 = {}, acc01 = {}, acc10 = {}, acc11 = {};
  for (int k0 = 0; k0 < K; k0 += 32) {
    if (k0 + 32 < K) {
      __builtin_prefetch(Ar0 + k0 + 32, 0, 0);   // global_prefetch_b8
      __builtin_prefetch(Ar1 + k0 + 32, 0, 0);
    }
    const v16bf a0 = pack_a(Ar0 + k0);
    const v16bf a1 = pack_a(Ar1 + k0);
    const int krow = k0 + half * 8;
    const v16bf b0 = pack_b(W0, krow, N);
    const v16bf b1 = pack_b(W1, krow, N);
    acc00 = __builtin_amdgcn_wmma_f32_16x16x32_bf16(false, a0, false, b0, (short)0, acc00, false, false);
    acc01 = __builtin_amdgcn_wmma_f32_16x16x32_bf16(false, a0, false, b1, (short)0, acc01, false, false);
    acc10 = __builtin_amdgcn_wmma_f32_16x16x32_bf16(false, a1, false, b0, (short)0, acc10, false, false);
    acc11 = __builtin_amdgcn_wmma_f32_16x16x32_bf16(false, a1, false, b1, (short)0, acc11, false, false);
  }

  const int mb0 = tm * 32 + half * 8;
  const int mb1 = mb0 + 16;
#pragma unroll
  for (int sub = 0; sub < 4; ++sub) {
    const v8f acc = (sub == 0) ? acc00 : (sub == 1) ? acc01 : (sub == 2) ? acc10 : acc11;
    const int col = (sub & 1) ? col1 : col0;
    const int mb  = (sub & 2) ? mb1 : mb0;
    if (col < N) {
      const float bv = bias ? bias[col] : 0.f;
#pragma unroll
      for (int r = 0; r < 8; ++r) {
        float v = applyact<ACT>(acc[r] + bv);
        const size_t o = (size_t)(mb + r) * N + col;
        if (ADD_RES) v += res[o];
        C[o] = v;
      }
    }
  }
}

// ---------------------------------------------------------------- LayerNorm
__global__ void __launch_bounds__(256)
ln_split_kernel(const float* __restrict__ x, const float* __restrict__ w,
                const float* __restrict__ b, float* __restrict__ out_lo,
                float* __restrict__ out_hi, int rows)
{
  const int wave = threadIdx.x >> 5, lane = threadIdx.x & 31;
  const size_t row = (size_t)blockIdx.x * 8 + wave;
  if (row >= (size_t)rows) return;
  const float* xr = x + row * 128;
  float v[4]; float s = 0.f;
#pragma unroll
  for (int j = 0; j < 4; ++j) { v[j] = xr[lane + 32 * j]; s += v[j]; }
#pragma unroll
  for (int off = 16; off; off >>= 1) s += __shfl_xor(s, off, 32);
  const float mean = s * (1.f / 128.f);
  float s2 = 0.f;
#pragma unroll
  for (int j = 0; j < 4; ++j) { const float d = v[j] - mean; s2 += d * d; }
#pragma unroll
  for (int off = 16; off; off >>= 1) s2 += __shfl_xor(s2, off, 32);
  const float rstd = rsqrtf(s2 * (1.f / 128.f) + 1e-5f);
#pragma unroll
  for (int j = 0; j < 4; ++j) {
    const int c = lane + 32 * j;
    const float y = (v[j] - mean) * rstd * w[c] + b[c];
    if (c < 64) out_lo[row * 64 + c] = y; else out_hi[row * 64 + (c - 64)] = y;
  }
}

template<int C, bool RES>
__global__ void __launch_bounds__(256)
ln_kernel(const float* __restrict__ x, const float* __restrict__ xres,
          const float* __restrict__ w, const float* __restrict__ b,
          float* __restrict__ out, int rows)
{
  constexpr int PER = C / 32;
  const int wave = threadIdx.x >> 5, lane = threadIdx.x & 31;
  const size_t row = (size_t)blockIdx.x * 8 + wave;
  if (row >= (size_t)rows) return;
  const float* xr = x + row * C;
  float v[PER]; float s = 0.f;
#pragma unroll
  for (int j = 0; j < PER; ++j) {
    const int c = lane + 32 * j;
    float t = xr[c];
    if (RES) t += xres[row * C + c];
    v[j] = t; s += t;
  }
#pragma unroll
  for (int off = 16; off; off >>= 1) s += __shfl_xor(s, off, 32);
  const float mean = s * (1.f / C);
  float s2 = 0.f;
#pragma unroll
  for (int j = 0; j < PER; ++j) { const float d = v[j] - mean; s2 += d * d; }
#pragma unroll
  for (int off = 16; off; off >>= 1) s2 += __shfl_xor(s2, off, 32);
  const float rstd = rsqrtf(s2 * (1.f / C) + 1e-5f);
#pragma unroll
  for (int j = 0; j < PER; ++j) {
    const int c = lane + 32 * j;
    out[row * C + c] = (v[j] - mean) * rstd * w[c] + b[c];
  }
}

// ---------------------------------------------- NSS serpentine gather (closed form)
__global__ void __launch_bounds__(256)
gather_seqs_kernel(const float* __restrict__ xs, float* __restrict__ seqs)
{
  const size_t gid = (size_t)blockIdx.x * 256 + threadIdx.x;  // 32*4096*64
  if (gid >= (size_t)32 * 4096 * 64) return;
  const int ch = gid & 63;
  const int p  = (gid >> 6) & 4095;
  const int db = (int)(gid >> 18);
  const int d = db >> 3, bb = db & 7;
  const int pe = (d & 1) ? (4095 - p) : p;
  const int s = pe >> 8, q = pe & 255, lr = q >> 6, j = q & 63;
  int idx;
  if (d < 2) { const int r = s * 4 + lr; const int c = (lr & 1) ? (63 - j) : j; idx = r * 64 + c; }
  else       { const int c = s * 4 + lr; const int r = (lr & 1) ? (63 - j) : j; idx = r * 64 + c; }
  seqs[gid] = xs[((size_t)bb * 4096 + idx) * 64 + ch];
}

__global__ void __launch_bounds__(256)
unscatter_kernel(const float* __restrict__ mo, const float* __restrict__ xm_in,
                 float* __restrict__ spatial)
{
  const size_t gid = (size_t)blockIdx.x * 256 + threadIdx.x;  // 8*4096*64
  if (gid >= (size_t)8 * 4096 * 64) return;
  const int ch = gid & 63;
  const int l  = (gid >> 6) & 4095;
  const int bb = (int)(gid >> 18);
  const int r = l >> 6, cc = l & 63;
  const int sh = r >> 2, lr = r & 3;
  const int ph = sh * 256 + lr * 64 + ((lr & 1) ? (63 - cc) : cc);
  const int sv = cc >> 2, lc = cc & 3;
  const int pv = sv * 256 + lc * 64 + ((lc & 1) ? (63 - r) : r);
  float acc =
      mo[(((size_t)(0 * 8 + bb)) * 4096 + ph)          * 64 + ch] +
      mo[(((size_t)(1 * 8 + bb)) * 4096 + (4095 - ph)) * 64 + ch] +
      mo[(((size_t)(2 * 8 + bb)) * 4096 + pv)          * 64 + ch] +
      mo[(((size_t)(3 * 8 + bb)) * 4096 + (4095 - pv)) * 64 + ch];
  spatial[gid] = 0.25f * acc + xm_in[gid];
}

// -------------------------------------------------- depthwise causal conv + silu
__global__ void __launch_bounds__(256)
conv_silu_kernel(const float* __restrict__ zx, const float* __restrict__ cw,
                 const float* __restrict__ cb, float* __restrict__ xBC)
{
  const size_t gid = (size_t)blockIdx.x * 256 + threadIdx.x;  // 32*4096*256
  if (gid >= (size_t)32 * 4096 * 256) return;
  const int c  = gid & 255;
  const int l  = (gid >> 8) & 4095;
  const int bb = (int)(gid >> 20);
  const float* base = zx + ((size_t)bb * 4096) * DINPROJ + 128 + c;
  float acc = cb[c];
#pragma unroll
  for (int j = 0; j < 4; ++j) {
    const int ll = l - 3 + j;
    if (ll >= 0) acc += cw[c * 4 + j] * base[(size_t)ll * DINPROJ];
  }
  xBC[gid] = acc / (1.f + __expf(-acc));
}

// ------------------------------------------------------ sequential SSM scan
__global__ void __launch_bounds__(256)
scan_kernel(const float* __restrict__ zx, const float* __restrict__ xBC,
            const float* __restrict__ dt_bias, const float* __restrict__ A_log,
            const float* __restrict__ Dp, float* __restrict__ y)
{
  const int bb = blockIdx.x >> 1;
  const int hh = blockIdx.x & 1;
  const int t  = threadIdx.x;
  const int p  = t >> 2;        // headdim index 0..63
  const int q  = t & 3;
  const int n0 = q * 16;        // 16 state columns per thread
  float hstate[16];
#pragma unroll
  for (int i = 0; i < 16; ++i) hstate[i] = 0.f;
  __shared__ float sx[64], sB[64], sC[64], sdt[2];
  const float Acoef = -__expf(A_log[hh]);
  const float dtb = dt_bias[hh];
  const float Dh  = Dp[hh];
  const float* xrow  = xBC + ((size_t)bb * 4096) * 256;
  const float* dtrow = zx  + ((size_t)bb * 4096) * DINPROJ + 384 + hh;
  float* yrow = y + ((size_t)bb * 4096) * 128 + hh * 64;

  for (int l = 0; l < 4096; ++l) {
    const float* xb = xrow + (size_t)l * 256;
#if defined(HAS_ASYNC32)
    if (t < 64)        async_ld_b32(xb + hh * 64 + t,    &sx[t]);
    else if (t < 128)  async_ld_b32(xb + 128 + (t - 64), &sB[t - 64]);
    else if (t < 192)  async_ld_b32(xb + 192 + (t - 128),&sC[t - 128]);
    else if (t == 192) {
      const float dtr = dtrow[(size_t)l * DINPROJ] + dtb;
      const float dtv = (dtr > 20.f) ? dtr : log1pf(__expf(dtr));
      sdt[0] = dtv;
      sdt[1] = __expf(Acoef * dtv);
    }
    wait_async0();
#else
    if (t < 64)        sx[t]       = xb[hh * 64 + t];
    else if (t < 128)  sB[t - 64]  = xb[128 + (t - 64)];
    else if (t < 192)  sC[t - 128] = xb[192 + (t - 128)];
    else if (t == 192) {
      const float dtr = dtrow[(size_t)l * DINPROJ] + dtb;
      const float dtv = (dtr > 20.f) ? dtr : log1pf(__expf(dtr));
      sdt[0] = dtv;
      sdt[1] = __expf(Acoef * dtv);
    }
#endif
    __syncthreads();
    const float dtv = sdt[0], dA = sdt[1];
    const float xv = sx[p];
    const float coeff = dtv * xv;
    float ypart = 0.f;
#pragma unroll
    for (int i = 0; i < 16; ++i) {
      const float hn = hstate[i] * dA + coeff * sB[n0 + i];
      hstate[i] = hn;
      ypart += hn * sC[n0 + i];
    }
    ypart += __shfl_xor(ypart, 1, 32);
    ypart += __shfl_xor(ypart, 2, 32);
    if (q == 0) yrow[(size_t)l * 128 + p] = ypart + Dh * xv;
    __syncthreads();
  }
}

// ----------------------------------------------- gated RMSNorm: u = y*silu(z)
__global__ void __launch_bounds__(256)
gate_rms_kernel(float* __restrict__ y, const float* __restrict__ zx,
                const float* __restrict__ rms_w)
{
  const int wave = threadIdx.x >> 5, lane = threadIdx.x & 31;
  const size_t row = (size_t)blockIdx.x * 8 + wave;   // 131072 rows
  if (row >= (size_t)NSEQ) return;
  float* yr = y + row * 128;
  const float* zr = zx + row * DINPROJ;               // z = cols [0,128)
  float u[4]; float s = 0.f;
#pragma unroll
  for (int j = 0; j < 4; ++j) {
    const int c = lane + 32 * j;
    const float z = zr[c];
    const float uu = yr[c] * (z / (1.f + __expf(-z)));
    u[j] = uu; s += uu * uu;
  }
#pragma unroll
  for (int off = 16; off; off >>= 1) s += __shfl_xor(s, off, 32);
  const float r = rsqrtf(s * (1.f / 128.f) + 1e-5f);
#pragma unroll
  for (int j = 0; j < 4; ++j) {
    const int c = lane + 32 * j;
    yr[c] = u[j] * r * rms_w[c];
  }
}

// ------------------------------------------ tmix concat: [spatial[b], spatial[b^4]]
__global__ void __launch_bounds__(256)
build_cat_kernel(const float* __restrict__ spatial, float* __restrict__ cat)
{
  const size_t gid = (size_t)blockIdx.x * 256 + threadIdx.x;  // 8*4096*128
  if (gid >= (size_t)8 * 4096 * 128) return;
  const int ch = gid & 127;
  const int l  = (gid >> 7) & 4095;
  const int bb = (int)(gid >> 19);
  cat[gid] = (ch < 64)
      ? spatial[((size_t)bb * 4096 + l) * 64 + ch]
      : spatial[((size_t)(bb ^ 4) * 4096 + l) * 64 + (ch - 64)];
}

// --------------------------------------------------- 8x8 window attention
__global__ void __launch_bounds__(256)
wattn_kernel(const float* __restrict__ qkv, const float* __restrict__ xa_in,
             const float* __restrict__ gate_w, const float* __restrict__ gate_b,
             float* __restrict__ og)
{
  __shared__ __align__(16) float sqkv[64][192];   // 48 KB
  const int img = blockIdx.x >> 6;
  const int wi  = blockIdx.x & 63;
  const int wr = wi >> 3, wc = wi & 7;
  const int t = threadIdx.x;
#if defined(HAS_ASYNC128)
  for (int e = t; e < 64 * 48; e += 256) {   // 3072 x float4
    const int n = e / 48, m4 = e - n * 48;
    const int r = wr * 8 + (n >> 3), c = wc * 8 + (n & 7);
    async_ld_b128(&qkv[((size_t)img * 4096 + r * 64 + c) * 192 + m4 * 4],
                  &sqkv[n][m4 * 4]);
  }
  wait_async0();
#else
  for (int e = t; e < 64 * 192; e += 256) {
    const int n = e / 192, m = e - n * 192;
    const int r = wr * 8 + (n >> 3), c = wc * 8 + (n & 7);
    sqkv[n][m] = qkv[((size_t)img * 4096 + r * 64 + c) * 192 + m];
  }
#endif
  __syncthreads();
  const int h = t >> 6, qi = t & 63;
  float sc[64];
  float mx = -1e30f;
#pragma unroll 4
  for (int k = 0; k < 64; ++k) {
    float s = 0.f;
#pragma unroll
    for (int d = 0; d < 16; ++d) s += sqkv[qi][h * 16 + d] * sqkv[k][64 + h * 16 + d];
    s *= 0.25f;   // hd^-0.5, hd=16
    sc[k] = s; mx = fmaxf(mx, s);
  }
  float sum = 0.f;
#pragma unroll 4
  for (int k = 0; k < 64; ++k) { sc[k] = __expf(sc[k] - mx); sum += sc[k]; }
  const float inv = 1.f / sum;
  const int r = wr * 8 + (qi >> 3), c = wc * 8 + (qi & 7);
  const size_t l = (size_t)img * 4096 + r * 64 + c;
  const float* xr = xa_in + l * 64;
  float g = gate_b[h];
  for (int cc = 0; cc < 64; ++cc) g += xr[cc] * gate_w[cc * 4 + h];
  g = 1.f / (1.f + __expf(-g));
  float* ogr = og + l * 64 + h * 16;
#pragma unroll
  for (int d = 0; d < 16; ++d) {
    float o = 0.f;
#pragma unroll 4
    for (int k = 0; k < 64; ++k) o += sc[k] * sqkv[k][128 + h * 16 + d];
    ogr[d] = o * inv * g;
  }
}

// -------------------------------------------------------------- fusion / tail
__global__ void __launch_bounds__(256)
fuse_kernel(const float* __restrict__ xm, const float* __restrict__ xa,
            const float* __restrict__ g1, const float* __restrict__ g2,
            float* __restrict__ fus, size_t n)
{
  const size_t i = (size_t)blockIdx.x * 256 + threadIdx.x;
  if (i < n) fus[i] = xm[i] * g2[i] + xa[i] * g1[i];
}

__global__ void __launch_bounds__(256)
pool_kernel(const float* __restrict__ y2d, float* __restrict__ pool)
{
  __shared__ float sp[256];
  const int b = blockIdx.x, t = threadIdx.x;
  const int ch = t & 127, hfl = t >> 7;
  float s = 0.f;
  for (int l = hfl; l < 4096; l += 2) s += y2d[((size_t)b * 4096 + l) * 128 + ch];
  sp[t] = s;
  __syncthreads();
  if (t < 128) pool[b * 128 + ch] = (sp[t] + sp[t + 128]) * (1.f / 4096.f);
}

__global__ void __launch_bounds__(128)
catt_kernel(const float* __restrict__ pool, const float* __restrict__ w1,
            const float* __restrict__ b1, const float* __restrict__ w2,
            const float* __restrict__ b2, float* __restrict__ catt)
{
  __shared__ float sp[128], sh[32];
  const int b = blockIdx.x, t = threadIdx.x;
  sp[t] = pool[b * 128 + t];
  __syncthreads();
  if (t < 32) {
    float a = b1[t];
    for (int c = 0; c < 128; ++c) a += sp[c] * w1[c * 32 + t];
    sh[t] = fmaxf(a, 0.f);
  }
  __syncthreads();
  float a = b2[t];
  for (int j = 0; j < 32; ++j) a += sh[j] * w2[j * 128 + t];
  catt[b * 128 + t] = 1.f / (1.f + __expf(-a));
}

__global__ void __launch_bounds__(256)
xout_kernel(const float* __restrict__ x, const float* __restrict__ y2d,
            const float* __restrict__ catt, float* __restrict__ xout)
{
  const size_t i = (size_t)blockIdx.x * 256 + threadIdx.x;  // 8*4096*128
  if (i >= (size_t)8 * 4096 * 128) return;
  const int ch = i & 127;
  const int b  = (int)(i >> 19);
  xout[i] = x[i] + y2d[i] * catt[b * 128 + ch];
}

// =========================================================== host launcher
template<int ACT, bool ADD_RES>
static inline void launch_gemm(const float* A, const float* W, const float* bias,
                               const float* res, float* C, int M, int N, int K,
                               hipStream_t s)
{
  const int tiles = (M >> 5) * ((N + 31) >> 5);
  gemm_bf16_wmma<ACT, ADD_RES><<<(tiles + 3) >> 2, 128, 0, s>>>(A, W, bias, res, C, M, N, K);
}

extern "C" void kernel_launch(void* const* d_in, const int* in_sizes, int n_in,
                              void* d_out, int out_size, void* d_ws, size_t ws_size,
                              hipStream_t stream)
{
  (void)in_sizes; (void)n_in; (void)out_size; (void)ws_size;
  const float* x         = (const float*)d_in[0];
  const float* norm1_w   = (const float*)d_in[3];
  const float* norm1_b   = (const float*)d_in[4];
  const float* norm_s_w  = (const float*)d_in[5];
  const float* norm_s_b  = (const float*)d_in[6];
  const float* norm_t_w  = (const float*)d_in[7];
  const float* norm_t_b  = (const float*)d_in[8];
  const float* tm_w1     = (const float*)d_in[9];
  const float* tm_b1     = (const float*)d_in[10];
  const float* tm_w2     = (const float*)d_in[11];
  const float* tm_b2     = (const float*)d_in[12];
  const float* in_proj_w = (const float*)d_in[13];
  const float* conv_w    = (const float*)d_in[14];
  const float* conv_b    = (const float*)d_in[15];
  const float* dt_bias   = (const float*)d_in[16];
  const float* A_log     = (const float*)d_in[17];
  const float* Dp        = (const float*)d_in[18];
  const float* rms_w     = (const float*)d_in[19];
  const float* out_proj_w= (const float*)d_in[20];
  const float* qkv_w     = (const float*)d_in[21];
  const float* qkv_b     = (const float*)d_in[22];
  const float* proj_w    = (const float*)d_in[23];
  const float* proj_b    = (const float*)d_in[24];
  const float* gate_w    = (const float*)d_in[25];
  const float* gate_b    = (const float*)d_in[26];
  const float* cg_w1     = (const float*)d_in[27];
  const float* cg_b1     = (const float*)d_in[28];
  const float* cg_w2     = (const float*)d_in[29];
  const float* cg_b2     = (const float*)d_in[30];
  const float* op_w      = (const float*)d_in[31];
  const float* op_b      = (const float*)d_in[32];
  const float* cab_w1    = (const float*)d_in[33];
  const float* cab_b1    = (const float*)d_in[34];
  const float* cab_w2    = (const float*)d_in[35];
  const float* cab_b2    = (const float*)d_in[36];
  const float* norm2_w   = (const float*)d_in[37];
  const float* norm2_b   = (const float*)d_in[38];
  const float* mlp_w1    = (const float*)d_in[39];
  const float* mlp_b1    = (const float*)d_in[40];
  const float* mlp_w2    = (const float*)d_in[41];
  const float* mlp_b2    = (const float*)d_in[42];
  float* out = (float*)d_out;
  float* ws  = (float*)d_ws;

  // ---- workspace layout (floats), with lifetime-based reuse ----
  constexpr size_t U = (size_t)NTOK * 64;               // 2,097,152
  constexpr size_t ZX_SZ = (size_t)NSEQ * DINPROJ;      // 50,593,792
  const size_t OFF_XM_IN = 0;
  const size_t OFF_XA_IN = OFF_XM_IN + U;
  const size_t OFF_XS    = OFF_XA_IN + U;               // reused: TMH
  const size_t OFF_SEQS  = OFF_XS + U;                  // 4U, reused: MO
  const size_t OFF_ZX    = OFF_SEQS + 4 * U;            // reused: Y2D/XOUT/H2
  const size_t OFF_XBC   = OFF_ZX + ZX_SZ;              // 16U, reused: attn/gate bufs
  const size_t OFF_YB    = OFF_XBC + 16 * U;            // 8U, reused: HID
  const size_t OFF_SPAT  = OFF_YB + 8 * U;
  const size_t OFF_POOL  = OFF_SPAT + U;
  const size_t OFF_CATT  = OFF_POOL + 1024;
  // aliases
  const size_t OFF_TMH = OFF_XS;
  const size_t OFF_MO  = OFF_SEQS;
  const size_t OFF_Y2D = OFF_ZX;
  const size_t OFF_XOUT= OFF_ZX + 2 * U;
  const size_t OFF_H2  = OFF_ZX + 4 * U;
  const size_t OFF_CAT = OFF_XBC;
  const size_t OFF_TMO = OFF_XBC + 2 * U;
  const size_t OFF_XM  = OFF_XBC + 3 * U;
  const size_t OFF_QKV = OFF_XBC + 4 * U;   // 3U
  const size_t OFF_OG  = OFF_XBC + 7 * U;
  const size_t OFF_XA  = OFF_XBC + 8 * U;
  const size_t OFF_G1  = OFF_XBC + 9 * U;
  const size_t OFF_G2  = OFF_XBC + 10 * U;
  const size_t OFF_FUS = OFF_XBC + 11 * U;
  const size_t OFF_HID = OFF_YB;

  // 1) norm1 + channel split
  ln_split_kernel<<<NTOK / 8, 256, 0, stream>>>(x, norm1_w, norm1_b,
      ws + OFF_XM_IN, ws + OFF_XA_IN, NTOK);
  // 2) norm_s
  ln_kernel<64, false><<<NTOK / 8, 256, 0, stream>>>(ws + OFF_XM_IN, nullptr,
      norm_s_w, norm_s_b, ws + OFF_XS, NTOK);
  // 3) serpentine gather into 4-direction sequences
  gather_seqs_kernel<<<(32u * 4096 * 64) / 256, 256, 0, stream>>>(ws + OFF_XS, ws + OFF_SEQS);
  // 4) in_proj GEMM: (131072x64)@(64x386)
  launch_gemm<0, false>(ws + OFF_SEQS, in_proj_w, nullptr, nullptr,
                        ws + OFF_ZX, NSEQ, DINPROJ, 64, stream);
  // 5) depthwise causal conv + silu on xBC slice
  conv_silu_kernel<<<(32u * 4096 * 256) / 256, 256, 0, stream>>>(ws + OFF_ZX,
      conv_w, conv_b, ws + OFF_XBC);
  // 6) sequential SSM scan
  scan_kernel<<<64, 256, 0, stream>>>(ws + OFF_ZX, ws + OFF_XBC,
      dt_bias, A_log, Dp, ws + OFF_YB);
  // 7) gated RMSNorm (in place on y)
  gate_rms_kernel<<<NSEQ / 8, 256, 0, stream>>>(ws + OFF_YB, ws + OFF_ZX, rms_w);
  // 8) out_proj GEMM: (131072x128)@(128x64)
  launch_gemm<0, false>(ws + OFF_YB, out_proj_w, nullptr, nullptr,
                        ws + OFF_MO, NSEQ, 64, 128, stream);
  // 9) inverse scatter + average + residual
  unscatter_kernel<<<(8u * 4096 * 64) / 256, 256, 0, stream>>>(ws + OFF_MO,
      ws + OFF_XM_IN, ws + OFF_SPAT);
  // 10) tmix concat
  build_cat_kernel<<<(8u * 4096 * 128) / 256, 256, 0, stream>>>(ws + OFF_SPAT, ws + OFF_CAT);
  // 11-12) tmix GEMMs (silu hidden)
  launch_gemm<1, false>(ws + OFF_CAT, tm_w1, tm_b1, nullptr,
                        ws + OFF_TMH, NTOK, 64, 128, stream);
  launch_gemm<0, false>(ws + OFF_TMH, tm_w2, tm_b2, nullptr,
                        ws + OFF_TMO, NTOK, 64, 64, stream);
  // 13) norm_t with fused residual
  ln_kernel<64, true><<<NTOK / 8, 256, 0, stream>>>(ws + OFF_SPAT, ws + OFF_TMO,
      norm_t_w, norm_t_b, ws + OFF_XM, NTOK);
  // 14) qkv GEMM: (32768x64)@(64x192)
  launch_gemm<0, false>(ws + OFF_XA_IN, qkv_w, qkv_b, nullptr,
                        ws + OFF_QKV, NTOK, 192, 64, stream);
  // 15) window attention + gate
  wattn_kernel<<<8 * 64, 256, 0, stream>>>(ws + OFF_QKV, ws + OFF_XA_IN,
      gate_w, gate_b, ws + OFF_OG);
  // 16) attention proj
  launch_gemm<0, false>(ws + OFF_OG, proj_w, proj_b, nullptr,
                        ws + OFF_XA, NTOK, 64, 64, stream);
  // 17-18) cross gates (sigmoid)
  launch_gemm<3, false>(ws + OFF_XA, cg_w2, cg_b2, nullptr,
                        ws + OFF_G2, NTOK, 64, 64, stream);
  launch_gemm<3, false>(ws + OFF_XM, cg_w1, cg_b1, nullptr,
                        ws + OFF_G1, NTOK, 64, 64, stream);
  // 19) fuse
  fuse_kernel<<<(unsigned)((U + 255) / 256), 256, 0, stream>>>(ws + OFF_XM,
      ws + OFF_XA, ws + OFF_G1, ws + OFF_G2, ws + OFF_FUS, U);
  // 20) output proj: (32768x64)@(64x128)
  launch_gemm<0, false>(ws + OFF_FUS, op_w, op_b, nullptr,
                        ws + OFF_Y2D, NTOK, 128, 64, stream);
  // 21-22) channel attention
  pool_kernel<<<8, 256, 0, stream>>>(ws + OFF_Y2D, ws + OFF_POOL);
  catt_kernel<<<8, 128, 0, stream>>>(ws + OFF_POOL, cab_w1, cab_b1, cab_w2,
      cab_b2, ws + OFF_CATT);
  // 23) xout = shortcut + y2d*catt
  xout_kernel<<<(8u * 4096 * 128) / 256, 256, 0, stream>>>(x, ws + OFF_Y2D,
      ws + OFF_CATT, ws + OFF_XOUT);
  // 24) norm2
  ln_kernel<128, false><<<NTOK / 8, 256, 0, stream>>>(ws + OFF_XOUT, nullptr,
      norm2_w, norm2_b, ws + OFF_H2, NTOK);
  // 25) MLP fc1 + exact gelu: (32768x128)@(128x512)
  launch_gemm<2, false>(ws + OFF_H2, mlp_w1, mlp_b1, nullptr,
                        ws + OFF_HID, NTOK, 512, 128, stream);
  // 26) MLP fc2 + residual, straight into d_out: (32768x512)@(512x128)
  launch_gemm<0, true>(ws + OFF_HID, mlp_w2, mlp_b2, ws + OFF_XOUT,
                       out, NTOK, 128, 512, stream);
}
